// CodebookEMA_9560597201279
// MI455X (gfx1250) — compile-verified, matrix operand order checked by reading, not run on hardware
//
#include <hip/hip_runtime.h>
#include <hip/hip_bf16.h>

typedef __attribute__((ext_vector_type(16))) __bf16 v16bf;
typedef __attribute__((ext_vector_type(8)))  float  v8f;

#define OFF_LOSS 16777216
#define OFF_PERP 16777217
#define OFF_IDX  16777218

__device__ __forceinline__ unsigned short f2bf(float f) {
    unsigned int u = __float_as_uint(f);
    unsigned int r = u + 0x7FFFu + ((u >> 16) & 1u);
    return (unsigned short)(r >> 16);
}

// 16-bit A/B fragment layout (ISA 7.12.2): lane group g covers K {8g..8g+7, 16+8g..16+8g+7}
__device__ __forceinline__ void frag_pos(int kk, int mn, int& lane, int& pos) {
    int q = kk >> 3;
    lane = mn + ((q & 1) << 4);
    pos  = ((q >> 1) << 3) + (kk & 7);
}

// 8x GLOBAL_LOAD_ASYNC_TO_LDS_B128 covering 32KB (256 threads x 16B x 8 slices);
// INST_OFFSET is added to BOTH the LDS and global address (ISA 08 §4.4).
__device__ __forceinline__ void async8(unsigned lds, unsigned long long g) {
    asm volatile("global_load_async_to_lds_b128 %0, %1, off"              :: "v"(lds), "v"(g) : "memory");
    asm volatile("global_load_async_to_lds_b128 %0, %1, off offset:%c2"   :: "v"(lds), "v"(g), "n"(4096)  : "memory");
    asm volatile("global_load_async_to_lds_b128 %0, %1, off offset:%c2"   :: "v"(lds), "v"(g), "n"(8192)  : "memory");
    asm volatile("global_load_async_to_lds_b128 %0, %1, off offset:%c2"   :: "v"(lds), "v"(g), "n"(12288) : "memory");
    asm volatile("global_load_async_to_lds_b128 %0, %1, off offset:%c2"   :: "v"(lds), "v"(g), "n"(16384) : "memory");
    asm volatile("global_load_async_to_lds_b128 %0, %1, off offset:%c2"   :: "v"(lds), "v"(g), "n"(20480) : "memory");
    asm volatile("global_load_async_to_lds_b128 %0, %1, off offset:%c2"   :: "v"(lds), "v"(g), "n"(24576) : "memory");
    asm volatile("global_load_async_to_lds_b128 %0, %1, off offset:%c2"   :: "v"(lds), "v"(g), "n"(28672) : "memory");
}

__device__ __forceinline__ void wait_async0() {
#if __has_builtin(__builtin_amdgcn_s_wait_asynccnt)
    __builtin_amdgcn_s_wait_asynccnt(0);
#else
    asm volatile("s_wait_asynccnt 0x0" ::: "memory");
#endif
}
__device__ __forceinline__ void wait_async16() {
#if __has_builtin(__builtin_amdgcn_s_wait_asynccnt)
    __builtin_amdgcn_s_wait_asynccnt(16);
#else
    asm volatile("s_wait_asynccnt 0x10" ::: "memory");
#endif
}

__global__ void vq_zero(float* scal) {
    int i = blockIdx.x * 256 + threadIdx.x;
    if (i < 2112) scal[i] = 0.0f;   // loss(1)+pad(63)+w2(1024)+counts(1024)
}

// codebook fp32 -> bf16 hi/lo in WMMA B-fragment layout [ct(64)][ks(8)][lane(32)][16]
__global__ void vq_wfrag(const float* __restrict__ cb,
                         unsigned short* __restrict__ wfh,
                         unsigned short* __restrict__ wfl) {
    int i = blockIdx.x * 256 + threadIdx.x;    // 1024 blocks -> 262144 elements
    int k = i >> 8, c = i & 255;
    float f = cb[i];
    unsigned short hb = f2bf(f);
    float lr = f - __uint_as_float((unsigned int)hb << 16);
    unsigned short lb = f2bf(lr);
    int ct = k >> 4, n = k & 15;
    int ks = c >> 5, kk = c & 31;
    int ln, pos; frag_pos(kk, n, ln, pos);
    int off = ((ct * 8 + ks) * 32 + ln) * 16 + pos;
    wfh[off] = hb; wfl[off] = lb;
}

__global__ void vq_w2(const float* __restrict__ cb, float* __restrict__ w2g) {
    int k = blockIdx.x * 256 + threadIdx.x;   // 4 blocks -> 1024 codes
    const float* row = cb + (size_t)k * 256;
    float s = 0.0f;
    for (int c = 0; c < 256; ++c) { float v = row[c]; s += v * v; }
    w2g[k] = s;
}

__global__ __launch_bounds__(256, 1) void vq_main(
    const float* __restrict__ x, const float* __restrict__ cb,
    const unsigned short* __restrict__ wfh, const unsigned short* __restrict__ wfl,
    const float* __restrict__ w2g, float* __restrict__ out,
    float* __restrict__ loss_sum, unsigned int* __restrict__ counts)
{
    extern __shared__ __align__(16) char smem[];
    unsigned short* xhi = (unsigned short*)smem;        // 4rt*8ks*32*16 = 16384 (32KB)
    unsigned short* xlo = xhi + 16384;
    unsigned short* wb  = xlo + 16384;                  // 2 buffers x (hi 16384 + lo 16384) = 128KB
    float* redv = (float*)(wb + 65536);                 // 8 waves * 64 rows
    int*   redi = (int*)(redv + 512);
    int*   kidx = redi + 512;                           // 64
    float* zq   = (float*)smem;                         // reused after compute, 64*257 f32

    const int t    = threadIdx.x;
    const int lane = t & 31;
    const int wv   = t >> 5;
    const int ctl  = wv >> 1;            // this wave's code-tile within a 64-code chunk
    const int rtb  = (wv & 1) * 2;       // this wave's row-tile pair {rtb, rtb+1}
    const int blk  = blockIdx.x;         // b*64 + h
    const int b    = blk >> 6;
    const int h    = blk & 63;

    // low 32 bits of generic LDS pointer == LDS byte address (flat aperture truncation)
    const unsigned wbLds = (unsigned)(unsigned long long)(void*)wb + (unsigned)t * 16u;
    const unsigned long long gh0 = (unsigned long long)(const void*)wfh + (unsigned long long)t * 16ull;
    const unsigned long long gl0 = (unsigned long long)(const void*)wfl + (unsigned long long)t * 16ull;

    // ---- stage 64x256 x-block (NCHW gather) into bf16 hi/lo A-fragments
    const size_t xbase = ((size_t)b * 256) * 4096 + (size_t)h * 64;
    #pragma unroll 4
    for (int it = 0; it < 64; ++it) {
        int e = it * 256 + t;
        int w = e & 63, c = e >> 6;
        float f = x[xbase + (size_t)c * 4096 + w];
        unsigned short hb = f2bf(f);
        float lr = f - __uint_as_float((unsigned int)hb << 16);
        unsigned short lb = f2bf(lr);
        int rt = w >> 4, m = w & 15;
        int ks = c >> 5, kk = c & 31;
        int ln, pos; frag_pos(kk, m, ln, pos);
        int off = ((rt * 8 + ks) * 32 + ln) * 16 + pos;
        xhi[off] = hb; xlo[off] = lb;
    }

    float minv[2][8]; int mini[2][8];
    #pragma unroll
    for (int r = 0; r < 2; ++r)
        #pragma unroll
        for (int v = 0; v < 8; ++v) { minv[r][v] = 3.4e38f; mini[r][v] = 0; }

    // ---- prologue: DMA chunk 0 into buffer 0 (hi 32KB + lo 32KB)
    async8(wbLds,          gh0);
    async8(wbLds + 32768u, gl0);

    for (int ch = 0; ch < 16; ++ch) {
        const int cur = ch & 1;
        __syncthreads();  // B1: compute(ch-1) fully retired -> buffer (1-cur) free for DMA(ch+1)
        if (ch + 1 < 16) {
            unsigned nl = wbLds + (unsigned)(1 - cur) * 65536u;
            unsigned long long go = (unsigned long long)(ch + 1) * 32768ull;
            async8(nl,          gh0 + go);
            async8(nl + 32768u, gl0 + go);
            wait_async16();   // in-order ASYNCcnt: <=16 outstanding => chunk ch landed
        } else {
            wait_async0();
        }
        __syncthreads();  // B2: chunk ch visible to all waves

        const unsigned short* wbh = wb + cur * 32768;   // hi fragments of this buffer
        v8f acc[2] = {};
        #pragma unroll
        for (int ks = 0; ks < 8; ++ks) {
            const v16bf bh = *(const v16bf*)(wbh +         ((ctl * 8 + ks) * 32 + lane) * 16);
            const v16bf bl = *(const v16bf*)(wbh + 16384 + ((ctl * 8 + ks) * 32 + lane) * 16);
            #pragma unroll
            for (int r = 0; r < 2; ++r) {
                const v16bf ah = *(const v16bf*)(xhi + (((rtb + r) * 8 + ks) * 32 + lane) * 16);
                const v16bf al = *(const v16bf*)(xlo + (((rtb + r) * 8 + ks) * 32 + lane) * 16);
                acc[r] = __builtin_amdgcn_wmma_f32_16x16x32_bf16(false, ah, false, bh, (short)0, acc[r], false, false);
                acc[r] = __builtin_amdgcn_wmma_f32_16x16x32_bf16(false, al, false, bh, (short)0, acc[r], false, false);
                acc[r] = __builtin_amdgcn_wmma_f32_16x16x32_bf16(false, ah, false, bl, (short)0, acc[r], false, false);
            }
        }
        int code = ch * 64 + ctl * 16 + (lane & 15);
        float w2v = w2g[code];
        #pragma unroll
        for (int r = 0; r < 2; ++r)
            #pragma unroll
            for (int v = 0; v < 8; ++v) {
                float d = w2v - 2.0f * acc[r][v];
                if (d < minv[r][v]) { minv[r][v] = d; mini[r][v] = code; }
            }
    }

    // ---- argmin reduce: across 16 lanes (n) then across the 4 waves covering each rt-pair
    int g = lane >> 4;
    #pragma unroll
    for (int r = 0; r < 2; ++r)
        #pragma unroll
        for (int v = 0; v < 8; ++v) {
            float bv = minv[r][v]; int bi = mini[r][v];
            #pragma unroll
            for (int off = 8; off >= 1; off >>= 1) {
                float ov = __shfl_xor(bv, off, 32);
                int   oi = __shfl_xor(bi, off, 32);
                if (ov < bv || (ov == bv && oi < bi)) { bv = ov; bi = oi; }
            }
            if ((lane & 15) == 0) {
                int row = (rtb + r) * 16 + g * 8 + v;
                redv[wv * 64 + row] = bv;
                redi[wv * 64 + row] = bi;
            }
        }
    __syncthreads();
    if (t < 64) {
        int p = t >> 5;                 // rt-pair of this row (rt = t>>4)
        float bv = 3.4e38f; int bi = 0;
        for (int w = 0; w < 4; ++w) {
            int wvi = w * 2 + p;
            float ov = redv[wvi * 64 + t]; int oi = redi[wvi * 64 + t];
            if (ov < bv || (ov == bv && oi < bi)) { bv = ov; bi = oi; }
        }
        kidx[t] = bi;
        out[(size_t)OFF_IDX + (size_t)blk * 64 + t] = (float)bi;
        atomicAdd(&counts[bi], 1u);
    }
    __syncthreads();

    // ---- fused gather + NCHW writeback + commitment loss
    for (int r = 0; r < 64; ++r)
        zq[r * 257 + t] = cb[(size_t)kidx[r] * 256 + t];
    __syncthreads();

    float lsum = 0.0f;
    #pragma unroll 4
    for (int it = 0; it < 64; ++it) {
        int e = it * 256 + t;
        int w = e & 63, c = e >> 6;
        float zv = zq[w * 257 + c];
        size_t gi = xbase + (size_t)c * 4096 + w;
        float xv = x[gi];
        out[gi] = zv;
        float df = zv - xv; lsum += df * df;
    }
    #pragma unroll
    for (int off = 16; off >= 1; off >>= 1) lsum += __shfl_xor(lsum, off, 32);
    if (lane == 0) redv[wv] = lsum;
    __syncthreads();
    if (t == 0) {
        float s = 0.0f;
        for (int w = 0; w < 8; ++w) s += redv[w];
        atomicAdd(loss_sum, s);
    }
}

__global__ void vq_final(const float* __restrict__ loss_sum,
                         const unsigned int* __restrict__ counts,
                         float* __restrict__ out) {
    __shared__ float sm[1024];
    int t = threadIdx.x;
    float p = (float)counts[t] * (1.0f / 65536.0f);
    sm[t] = p * logf(p + 1e-10f);
    __syncthreads();
    for (int s = 512; s >= 1; s >>= 1) {
        if (t < s) sm[t] += sm[t + s];
        __syncthreads();
    }
    if (t == 0) {
        out[OFF_PERP] = expf(-sm[0]);
        out[OFF_LOSS] = 0.25f * loss_sum[0] * (1.0f / 16777216.0f);
    }
}

extern "C" void kernel_launch(void* const* d_in, const int* in_sizes, int n_in,
                              void* d_out, int out_size, void* d_ws, size_t ws_size,
                              hipStream_t stream) {
    (void)in_sizes; (void)n_in; (void)out_size; (void)ws_size;
    const float* x  = (const float*)d_in[0];   // (16,256,64,64) f32 NCHW
    const float* cb = (const float*)d_in[1];   // (1024,256) f32
    float* out = (float*)d_out;

    // ws layout: [0, 512KB) frag_hi | [512KB, 1MB) frag_lo | [1MB, ...) scalars
    unsigned short* wfh = (unsigned short*)d_ws;
    unsigned short* wfl = (unsigned short*)((char*)d_ws + 524288);
    float* scal = (float*)((char*)d_ws + 1048576);
    float* loss_sum      = scal;                            // 1 f32
    float* w2g           = scal + 64;                       // 1024 f32
    unsigned int* counts = (unsigned int*)(scal + 1088);    // 1024 u32

    vq_zero <<<9,    256, 0, stream>>>(scal);
    vq_wfrag<<<1024, 256, 0, stream>>>(cb, wfh, wfl);
    vq_w2   <<<4,    256, 0, stream>>>(cb, w2g);
    vq_main <<<1024, 256, 200960, stream>>>(x, cb, wfh, wfl, w2g, out, loss_sum, counts);
    vq_final<<<1,   1024, 0, stream>>>(loss_sum, counts, out);
}
